// MEMBlock_76192719831240
// MI455X (gfx1250) — matrix operand order repaired and checked by reading, not compiled, not run on hardware
//
#include <hip/hip_runtime.h>
#include <hip/hip_bf16.h>

typedef __attribute__((ext_vector_type(16))) _Float16 v16h;
typedef __attribute__((ext_vector_type(8)))  _Float16 v8h;
typedef __attribute__((ext_vector_type(8)))  float    v8f;

#define Bk   8
#define Ck   256
#define Nk   8192
#define Gk   4
#define GSk  64
#define Hk   64
#define TH3  192
#define EPSk 1e-5f

// pre is stored in WMMA C-fragment layout:
//   pre[g][t][gate(3)][sw(4)][lane16][vgpr(8=batch)]  (f32)
// -> 1536 floats (6KB) per (g,t); a scan wave's C operand is one 32B load.
#define PRE_T_STRIDE 1536

// Opaque register pin: keeps converted weight fragments live across the loop.
#define PIN_V(x) asm volatile("" : "+v"(x))

// Fast reciprocal (v_rcp_f32) -- avoids IEEE div_scale/div_fmas expansion
// on the latency-critical scan path.
__device__ inline float fast_rcp(float x) { return __builtin_amdgcn_rcpf(x); }

// ---------------------------------------------------------------------------
// WMMA helpers (gfx1250, wave32).  16x16x32 f16 -> f32 accumulate.
// A as loaded here (16-bit, 16x32, contiguous k): lane L: row M = L&15; K = 16*half .. 16*half+15
// B as loaded here (16-bit, 32x16, contiguous k): lane L: col N = L&15; K = 16*half .. 16*half+15
// C/D layout (f32 16x16):  lane L: col N = L&15; VGPR v holds row M = v + 8*half
// ---------------------------------------------------------------------------

union AFragU { v16h v; struct { v8h lo, hi; } h; };

__device__ inline v16h load_a_frag_lds(const _Float16* row_base, int kbase, int half) {
  AFragU u;
  u.h.lo = *(const v8h*)(row_base + kbase + 16 * half);
  u.h.hi = *(const v8h*)(row_base + kbase + 16 * half + 8);
  return u.v;
}

__device__ inline v16h load_b_frag_f32(const float* wrow, int koff) {
  v16h r;
#pragma unroll
  for (int i = 0; i < 16; ++i) r[i] = (_Float16)wrow[koff + i];
  return r;
}

__device__ inline v8f wmma_16x16x32(v16h a, v16h b, v8f c) {
  return __builtin_amdgcn_wmma_f32_16x16x32_f16(false, a, false, b, (short)0, c,
                                                false, false);
}

// ---------------------------------------------------------------------------
// Kernel: sequential GRU scan.  One block per group, 4 waves; wave w owns h
// columns [16w,16w+16) of all three gates so gating stays in the C/D register
// layout.  h is double-buffered in LDS -> one barrier per step.  W_hh
// fragments are converted f32->f16 once and pinned in VGPRs.  Pre-gates
// (b_ih and the r/z parts of b_hh folded in) arrive in C-fragment layout:
// each WMMA C operand is a single 32B vector load, no shuffling.
// ---------------------------------------------------------------------------
__global__ __launch_bounds__(128) void gru_scan_kernel(
    const float* __restrict__ pre,   // [G][N][3][4][16][8] f32 (frag layout)
    const float* __restrict__ W_hh,  // [G,192,64]
    const float* __restrict__ b_hh,  // [G,192]
    _Float16* __restrict__ hcat)     // [B,N,256] f16
{
  __shared__ _Float16 hbuf[2][16][72];

  const int g    = blockIdx.x;
  const int tid  = threadIdx.x;
  const int lane = tid & 31;
  const int wv   = tid >> 5;
  const int l15  = lane & 15;
  const int half = lane >> 4;
  const int col  = wv * 16 + l15;    // h column 0..63 owned by this lane

  // resident W_hh B-fragments (f32 -> f16 once, pinned live for whole loop)
  const float* wr = W_hh + ((size_t)g * TH3 + 0   + col) * Hk;
  const float* wz = W_hh + ((size_t)g * TH3 + 64  + col) * Hk;
  const float* wn = W_hh + ((size_t)g * TH3 + 128 + col) * Hk;
  v16h br0 = load_b_frag_f32(wr, 16 * half), br1 = load_b_frag_f32(wr, 32 + 16 * half);
  v16h bz0 = load_b_frag_f32(wz, 16 * half), bz1 = load_b_frag_f32(wz, 32 + 16 * half);
  v16h bn0 = load_b_frag_f32(wn, 16 * half), bn1 = load_b_frag_f32(wn, 32 + 16 * half);
  PIN_V(br0); PIN_V(br1); PIN_V(bz0); PIN_V(bz1); PIN_V(bn0); PIN_V(bn1);

  // b_hh n-gate bias as a ready-made C operand (WMMA D!=C leaves it intact)
  float bhn = b_hh[g * TH3 + 128 + col];
  v8f bhnv;
#pragma unroll
  for (int v = 0; v < 8; ++v) bhnv[v] = bhn;
  PIN_V(bhnv);

  for (int i = tid; i < 2 * 16 * 72; i += 128)
    (&hbuf[0][0][0])[i] = (_Float16)0.f;

  float hprev[8];
#pragma unroll
  for (int v = 0; v < 8; ++v) hprev[v] = 0.f;

  const float* preg = pre + (size_t)g * Nk * PRE_T_STRIDE;
  __syncthreads();

  for (int t = 0; t < Nk; ++t) {
    const int rb = t & 1, wb = rb ^ 1;

    // prefetch next step's 6KB pre-gate block (global_prefetch_b8)
    if (t + 1 < Nk) {
      const float* pnx = preg + (size_t)(t + 1) * PRE_T_STRIDE + lane * 48;
      __builtin_prefetch(pnx, 0, 3);
    }

    // C operands straight from memory (frag layout, biases pre-folded)
    const float* pbase = preg + ((size_t)t * 12 + wv) * 128 + l15 * 8;
    v8f prv = *(const v8f*)(pbase);          // r gate  (gate stride = 512 f32)
    v8f pzv = *(const v8f*)(pbase + 512);    // z gate
    v8f pnv = *(const v8f*)(pbase + 1024);   // n gate (b_ih only)

    // A fragments from current h buffer
    const _Float16* hrow = &hbuf[rb][l15][0];
    v16h a0 = load_a_frag_lds(hrow, 0, half);
    v16h a1 = load_a_frag_lds(hrow, 32, half);

    v8f gr = wmma_16x16x32(a0, br0, prv);  gr = wmma_16x16x32(a1, br1, gr);
    v8f gz = wmma_16x16x32(a0, bz0, pzv);  gz = wmma_16x16x32(a1, bz1, gz);
    v8f gn = wmma_16x16x32(a0, bn0, bhnv); gn = wmma_16x16x32(a1, bn1, gn);

    // gating, state update, LDS store (next buffer).  All reciprocals use
    // v_rcp_f32 (fast) instead of IEEE division.
#pragma unroll
    for (int v = 0; v < 8; ++v) {
      float r  = fast_rcp(1.f + __expf(-gr[v]));
      float z  = fast_rcp(1.f + __expf(-gz[v]));
      float nx = pnv[v] + r * gn[v];
      float e2 = __expf(-2.f * nx);
      float nn = (1.f - e2) * fast_rcp(1.f + e2);   // tanh
      float hn = z * hprev[v] + (1.f - z) * nn;
      hprev[v] = hn;
      hbuf[wb][v + 8 * half][col] = (_Float16)hn;
    }

    // emit hcat[b, t, g*64+col] (valid rows live in lanes 0..15)
    if (lane < 16) {
#pragma unroll
      for (int v = 0; v < 8; ++v)
        hcat[((size_t)v * Nk + t) * Ck + g * Hk + col] = (_Float16)hprev[v];
    }
    __syncthreads();
  }
}

// ---------------------------------------------------------------------------
// Kernel: pre-gates.  pre = xg . W_ih^T + b_ih (+ b_hh for r/z gates),
// written directly in the scan's C-fragment layout.
// One block per (g, b, 16-n tile); x staged through LDS (transposed) so
// A-fragments come from two ds b128 loads per K-chunk.
// ---------------------------------------------------------------------------
__global__ __launch_bounds__(128) void pregate_kernel(
    const float* __restrict__ x,     // [B,C,N]
    const float* __restrict__ W_ih,  // [G,192,64]
    const float* __restrict__ b_ih,  // [G,192]
    const float* __restrict__ b_hh,  // [G,192]
    float* __restrict__ pre)         // [G][N][3][4][16][8] f32
{
  __shared__ _Float16 xs[16][72];

  const int blk  = blockIdx.x;
  const int nt   = blk % (Nk / 16);
  const int b    = (blk / (Nk / 16)) % Bk;
  const int g    = blk / ((Nk / 16) * Bk);
  const int n0   = nt * 16;
  const int tid  = threadIdx.x;
  const int lane = tid & 31;
  const int wv   = tid >> 5;     // wave 0..3 -> scan-wave slot
  const int l15  = lane & 15;
  const int half = lane >> 4;

  // stage x[b, g*64+k, n0+i] -> xs[i][k]  (coalesced 64B reads along n)
#pragma unroll
  for (int it = 0; it < 8; ++it) {
    int idx = tid + 128 * it;          // 0..1023
    int k = idx >> 4, i = idx & 15;
    float xv = x[((size_t)b * Ck + g * GSk + k) * Nk + n0 + i];
    xs[i][k] = (_Float16)xv;
  }
  __syncthreads();

  const _Float16* arow = &xs[l15][0];
  v16h a0 = load_a_frag_lds(arow, 0, half);
  v16h a1 = load_a_frag_lds(arow, 32, half);

#pragma unroll
  for (int t3 = 0; t3 < 3; ++t3) {           // gate: 0=r 1=z 2=n
    int j = t3 * 64 + wv * 16 + l15;
    const float* wrow = W_ih + ((size_t)g * TH3 + j) * GSk;
    v16h bf0 = load_b_frag_f32(wrow, 16 * half);
    v16h bf1 = load_b_frag_f32(wrow, 32 + 16 * half);
    float bias = b_ih[g * TH3 + j];
    if (t3 < 2) bias += b_hh[g * TH3 + j];   // fold r/z recurrent bias
    v8f acc;
#pragma unroll
    for (int v = 0; v < 8; ++v) acc[v] = bias;
    acc = wmma_16x16x32(a0, bf0, acc);
    acc = wmma_16x16x32(a1, bf1, acc);

    // scatter into fragment layout: t = n0 + v + 8*half, batch = fastest idx
    float* pf = pre + ((((size_t)g * Nk + n0) * 12 + t3 * 4 + wv) * 16 + l15) * 8 + b;
#pragma unroll
    for (int v = 0; v < 8; ++v)
      pf[(size_t)(v + 8 * half) * PRE_T_STRIDE] = acc[v];
  }
}

// ---------------------------------------------------------------------------
// Kernel: 1x1 conv (WMMA GEMM, K=256) fused with LayerNorm + ReLU + residual.
// One block per 16 rows (same batch, 16 consecutive n); output written as
// [B,C,N] with n-contiguous 64B stores.
// ---------------------------------------------------------------------------
__global__ __launch_bounds__(128) void proj_ln_kernel(
    const _Float16* __restrict__ hcat,  // [B,N,256]
    const float* __restrict__ Wp,       // [256,256]
    const float* __restrict__ bp,       // [256]
    const float* __restrict__ gamma,
    const float* __restrict__ beta,
    const float* __restrict__ x,        // [B,C,N] residual
    float* __restrict__ out)            // [B,C,N]
{
  __shared__ float ybuf[16][264];
  __shared__ float psum[16][8], psum2[16][8];
  __shared__ float smu[16], srs[16];

  const int row0 = blockIdx.x * 16;    // flattened (b, n)
  const int b    = row0 / Nk;
  const int n0   = row0 % Nk;
  const int tid  = threadIdx.x;
  const int lane = tid & 31;
  const int wv   = tid >> 5;
  const int l15  = lane & 15;
  const int half = lane >> 4;

  v8f acc[4];
#pragma unroll
  for (int jt = 0; jt < 4; ++jt) {
    float bias = bp[wv * 64 + jt * 16 + l15];
#pragma unroll
    for (int v = 0; v < 8; ++v) acc[jt][v] = bias;
  }

  const _Float16* arow = hcat + (size_t)row0 * Ck;
#pragma unroll 2
  for (int kc = 0; kc < 8; ++kc) {     // K = 256 in chunks of 32
    AFragU u;
    const _Float16* ap = arow + (size_t)l15 * Ck + kc * 32;
    u.h.lo = *(const v8h*)(ap + 16 * half);
    u.h.hi = *(const v8h*)(ap + 16 * half + 8);
    v16h a = u.v;
#pragma unroll
    for (int jt = 0; jt < 4; ++jt) {
      const float* wrow = Wp + (size_t)(wv * 64 + jt * 16 + l15) * Ck;
      v16h bfr = load_b_frag_f32(wrow, kc * 32 + 16 * half);
      acc[jt] = wmma_16x16x32(a, bfr, acc[jt]);
    }
  }

#pragma unroll
  for (int jt = 0; jt < 4; ++jt) {
    int j = wv * 64 + jt * 16 + l15;
#pragma unroll
    for (int v = 0; v < 8; ++v)
      ybuf[v + 8 * half][j] = acc[jt][v];
  }
  __syncthreads();

  // LayerNorm over 256 channels: 8 partial sums per row, then reduce
  const int r  = tid & 15;
  const int cg = tid >> 4;             // 0..7
  {
    float s = 0.f, s2 = 0.f;
#pragma unroll 4
    for (int cc = 0; cc < 32; ++cc) {
      float y = ybuf[r][cg * 32 + cc];
      s += y; s2 += y * y;
    }
    psum[r][cg] = s; psum2[r][cg] = s2;
  }
  __syncthreads();
  if (tid < 16) {
    float S = 0.f, S2 = 0.f;
#pragma unroll
    for (int i = 0; i < 8; ++i) { S += psum[tid][i]; S2 += psum2[tid][i]; }
    float mu  = S * (1.f / Ck);
    float var = S2 * (1.f / Ck) - mu * mu;
    smu[tid] = mu;
    srs[tid] = rsqrtf(var + EPSk);
  }
  __syncthreads();

  const float mu = smu[r], rs = srs[r];
#pragma unroll 4
  for (int cc = 0; cc < 32; ++cc) {
    int c = cg * 32 + cc;
    float y = (ybuf[r][c] - mu) * rs * gamma[c] + beta[c];
    y = fmaxf(y, 0.f);
    size_t oidx = ((size_t)b * Ck + c) * Nk + n0 + r;
    out[oidx] = y + x[oidx];             // residual, n-contiguous stores
  }
}

// ---------------------------------------------------------------------------
extern "C" void kernel_launch(void* const* d_in, const int* in_sizes, int n_in,
                              void* d_out, int out_size, void* d_ws, size_t ws_size,
                              hipStream_t stream) {
  const float* x      = (const float*)d_in[0];
  const float* W_ih   = (const float*)d_in[1];
  const float* W_hh   = (const float*)d_in[2];
  const float* b_ih   = (const float*)d_in[3];
  const float* b_hh   = (const float*)d_in[4];
  const float* W_proj = (const float*)d_in[5];
  const float* b_proj = (const float*)d_in[6];
  const float* gamma  = (const float*)d_in[7];
  const float* beta   = (const float*)d_in[8];
  float* out = (float*)d_out;

  float* pre = (float*)d_ws;                                   // 192 MB f32
  _Float16* hcat = (_Float16*)((char*)d_ws +
                   (size_t)Gk * Nk * PRE_T_STRIDE * sizeof(float)); // +32 MB f16

  pregate_kernel<<<Gk * Bk * (Nk / 16), 128, 0, stream>>>(x, W_ih, b_ih, b_hh, pre);
  gru_scan_kernel<<<Gk, 128, 0, stream>>>(pre, W_hh, b_hh, hcat);
  proj_ln_kernel<<<(Bk * Nk) / 16, 128, 0, stream>>>(hcat, W_proj, b_proj,
                                                     gamma, beta, x, out);
}
